// CoordNet_46514495816183
// MI455X (gfx1250) — matrix operand order repaired and checked by reading, not compiled, not run on hardware
//
#include <hip/hip_runtime.h>
#include <hip/hip_bf16.h>

typedef __attribute__((ext_vector_type(16))) _Float16 v16h;
typedef __attribute__((ext_vector_type(8)))  _Float16 v8h;
typedef __attribute__((ext_vector_type(8)))  float    v8f;
typedef __attribute__((ext_vector_type(2)))  float    v2f;

#define EMB   64
#define NB    256          // graphs
#define EPG   512          // edges per graph
#define ETOT  (NB*EPG)
#define KDIM  192          // 3*EMB
#define NTYPES 59
#define CHUNK 128
#define AP    200          // LDS pitch (halves) for edge-feature chunk
#define WPIT  200          // LDS pitch (halves) for WnT

#define SHUF16(lo,hi) __builtin_shufflevector(lo,hi,0,1,2,3,4,5,6,7,8,9,10,11,12,13,14,15)

// ---------------- kernel 1: per-graph edge MLP (WMMA f16) + M = p^T v + vR = M^T M /denom (WMMA f32) ----
__global__ __launch_bounds__(256)
void k_graph(const int* __restrict__ t, const float* __restrict__ c,
             const int* __restrict__ edges, const float* __restrict__ type_emb,
             const float* __restrict__ Wp, const float* __restrict__ bp,
             const float* __restrict__ Wn, const float* __restrict__ bn,
             float* __restrict__ featF)
{
    __shared__ __attribute__((aligned(16))) _Float16 wnt[EMB][WPIT];    // WnT (f16): [n][k]
    __shared__ __attribute__((aligned(16))) _Float16 Ach[CHUNK][AP];    // edge features (f16)
    __shared__ __attribute__((aligned(16))) _Float16 te16[NTYPES][EMB]; // type_emb (f16)
    __shared__ __attribute__((aligned(16))) float    pch[CHUNK][4];     // p per edge
    __shared__ float part[16][3][EMB];                                  // M partials (deterministic)
    __shared__ float dpart[CHUNK];
    __shared__ float Mlds[3][EMB];
    __shared__ float wp_s[3][EMB];
    __shared__ float bp_s[EMB];
    __shared__ float bn_s[EMB];
    __shared__ float denom_s;

    const int tid  = threadIdx.x;
    const int g    = blockIdx.x;
    const int w    = tid >> 5;
    const int lane = tid & 31;
    const int l15  = lane & 15;
    const bool hic = lane >= 16;

    // stage WnT + type_emb as f16, Wp/bp/bn as f32
    for (int i = tid; i < KDIM*EMB; i += 256) {
        int n = i & 63, k = i >> 6;
        wnt[n][k] = (_Float16)Wn[k*EMB + n];
    }
    for (int i = tid; i < NTYPES*EMB; i += 256)
        te16[0][i] = (_Float16)type_emb[i];
    for (int i = tid; i < 3*EMB; i += 256) wp_s[i>>6][i&63] = Wp[i];
    if (tid < EMB) { bp_s[tid] = bp[tid]; bn_s[tid] = bn[tid]; }
    __syncthreads();

    float Mp[4][3];
    #pragma unroll
    for (int nt = 0; nt < 4; nt++)
        for (int d = 0; d < 3; d++) Mp[nt][d] = 0.f;
    float dloc = 0.f;

    const int e_l     = tid >> 1;
    const int halfsel = tid & 1;

    for (int ci = 0; ci < 4; ci++) {
        // ---- stage 128 edges: [v_t[src] | v_p | v_t[tgt]] as f16 ----
        {
            int e  = g*EPG + ci*CHUNK + e_l;
            int s  = edges[e];
            int tg = edges[ETOT + e];
            float p0 = c[3*tg+0] - c[3*s+0];
            float p1 = c[3*tg+1] - c[3*s+1];
            float p2 = c[3*tg+2] - c[3*s+2];
            if (!halfsel) {
                pch[e_l][0] = p0; pch[e_l][1] = p1; pch[e_l][2] = p2;
                dloc += p0*p0 + p1*p1 + p2*p2;
                const v8h* te = (const v8h*)&te16[t[s]][0];
                #pragma unroll
                for (int j = 0; j < 8; j++)
                    *(v8h*)&Ach[e_l][j*8] = te[j];                    // v_t[src] (pure f16 copy)
                for (int j = 0; j < 32; j++)
                    Ach[e_l][EMB+j] = (_Float16)(p0*wp_s[0][j] + p1*wp_s[1][j] + p2*wp_s[2][j] + bp_s[j]);
            } else {
                for (int j = 32; j < 64; j++)
                    Ach[e_l][EMB+j] = (_Float16)(p0*wp_s[0][j] + p1*wp_s[1][j] + p2*wp_s[2][j] + bp_s[j]);
                const v8h* te = (const v8h*)&te16[t[tg]][0];
                #pragma unroll
                for (int j = 0; j < 8; j++)
                    *(v8h*)&Ach[e_l][2*EMB + j*8] = te[j];            // v_t[tgt]
            }
        }
        __syncthreads();

        // ---- WMMA GEMM: [128 x 192] (f16) x Wn [192 x 64] -> v chunk [128 x 64] (f32 accum) ----
        const int mrow = w*16 + l15;
        const int ab   = hic ? 8 : 0;     // A K-offset base (lanes 16-31 hold K 8..15 / 24..31)
        const int bb   = hic ? 8 : 0;     // B K base (lanes 16-31 hold K 8..15 / 24..31)
        v8f acc[4];
        #pragma unroll
        for (int nt = 0; nt < 4; nt++) acc[nt] = (v8f){0.f,0.f,0.f,0.f,0.f,0.f,0.f,0.f};

        #pragma unroll
        for (int kt = 0; kt < 6; kt++) {
            // issue ALL operand loads for this k-step first, then the 4 WMMAs back-to-back
            v8h alo = *(const v8h*)&Ach[mrow][kt*32 + ab];
            v8h ahi = *(const v8h*)&Ach[mrow][kt*32 + ab + 16];
            v16h bfr[4];
            #pragma unroll
            for (int nt = 0; nt < 4; nt++) {
                v8h blo = *(const v8h*)&wnt[nt*16 + l15][kt*32 + bb];
                v8h bhi = *(const v8h*)&wnt[nt*16 + l15][kt*32 + bb + 16];
                bfr[nt] = SHUF16(blo, bhi);
            }
            v16h a = SHUF16(alo, ahi);
            #pragma unroll
            for (int nt = 0; nt < 4; nt++)
                acc[nt] = __builtin_amdgcn_wmma_f32_16x16x32_f16(
                              false, a, false, bfr[nt], (short)0, acc[nt], false, false);
        }

        // ---- bias + accumulate M[d][n] += p[e][d] * v[e][n] ----
        float bbia[4];
        #pragma unroll
        for (int nt = 0; nt < 4; nt++) bbia[nt] = bn_s[nt*16 + l15];
        #pragma unroll
        for (int r = 0; r < 8; r++) {
            int el = w*16 + r + (hic ? 8 : 0);
            float q0 = pch[el][0], q1 = pch[el][1], q2 = pch[el][2];
            #pragma unroll
            for (int nt = 0; nt < 4; nt++) {
                float v = acc[nt][r] + bbia[nt];
                Mp[nt][0] += q0*v; Mp[nt][1] += q1*v; Mp[nt][2] += q2*v;
            }
        }
        __syncthreads();   // Ach/pch reused next chunk
    }

    // ---- deterministic reductions ----
    if (!halfsel) dpart[e_l] = dloc;
    {
        int slot = w*2 + (hic ? 1 : 0);
        #pragma unroll
        for (int nt = 0; nt < 4; nt++)
            for (int d = 0; d < 3; d++)
                part[slot][d][nt*16 + l15] = Mp[nt][d];
    }
    __syncthreads();
    if (tid < 192) {
        int d = tid >> 6, n = tid & 63;
        float s = 0.f;
        for (int sl = 0; sl < 16; sl++) s += part[sl][d][n];
        Mlds[d][n] = s;
    }
    if (tid == 0) {
        float s = 0.f;
        for (int i = 0; i < CHUNK; i++) s += dpart[i];
        denom_s = s;
    }
    __syncthreads();

    // ---- vR = M^T M / denom via V_WMMA_F32_16X16X4_F32 (K padded 3->4 with zeros) ----
    float inv = 1.0f / denom_s;
    #pragma unroll
    for (int ttile = 0; ttile < 2; ttile++) {
        int tile = w + ttile*8;
        int it = tile >> 2, jt = tile & 3;
        int mi = it*16 + l15, nj = jt*16 + l15;
        v2f a, b;
        a.x = hic ? Mlds[2][mi] : Mlds[0][mi];
        a.y = hic ? 0.f         : Mlds[1][mi];
        b.x = hic ? Mlds[2][nj] : Mlds[0][nj];
        b.y = hic ? 0.f         : Mlds[1][nj];
        v8f dacc = (v8f){0.f,0.f,0.f,0.f,0.f,0.f,0.f,0.f};
        dacc = __builtin_amdgcn_wmma_f32_16x16x4_f32(
                   false, a, false, b, (short)0, dacc, false, false);
        float* outg = featF + (size_t)g*4096;
        #pragma unroll
        for (int r = 0; r < 8; r++) {
            int i = it*16 + r + (hic ? 8 : 0);
            outg[i*64 + jt*16 + l15] = dacc[r] * inv;
        }
    }
}

// ---------------- W1 [4096x64] f32 -> W1T [64x4096] f16 ----------------
__global__ __launch_bounds__(256)
void k_w1t(const float* __restrict__ W1, _Float16* __restrict__ W1T)
{
    int i = blockIdx.x*256 + threadIdx.x;      // 262144 total
    int n = i >> 12;
    int k = i & 4095;
    W1T[(size_t)n*4096 + k] = (_Float16)W1[(size_t)k*64 + n];
}

// ---------------- features f32 -> f16 ----------------
__global__ __launch_bounds__(256)
void k_cvt(const float* __restrict__ f, _Float16* __restrict__ h)
{
    int i = blockIdx.x*256 + threadIdx.x;      // 1048576 total
    h[i] = (_Float16)f[i];
}

// ---------------- kernel 2: predictor. [256,4096]@[4096,64] WMMA, then small VALU tails ----
__global__ __launch_bounds__(256)
void k_pred(const _Float16* __restrict__ featH, const _Float16* __restrict__ W1T,
            const float* __restrict__ b1, const float* __restrict__ W2,
            const float* __restrict__ b2, const float* __restrict__ W3,
            const float* __restrict__ b3, float* __restrict__ out)
{
    __shared__ float h1s[64][65];
    __shared__ float h2s[64][65];
    const int tid  = threadIdx.x;
    const int w    = tid >> 5;
    const int lane = tid & 31;
    const int l15  = lane & 15;
    const bool hic = lane >= 16;
    const int g0   = blockIdx.x * 64;
    const int nt   = w & 3;
    const int mt0  = w >> 2;        // 0..1
    const int mt1  = mt0 + 2;       // 2..3
    const int ab   = hic ? 8 : 0;
    const int bb   = hic ? 8 : 0;

    const _Float16* rowA0 = featH + (size_t)(g0 + mt0*16 + l15) * 4096;
    const _Float16* rowA1 = featH + (size_t)(g0 + mt1*16 + l15) * 4096;
    const _Float16* rowB  = W1T   + (size_t)(nt*16 + l15) * 4096;

    v8f acc0 = (v8f){0.f,0.f,0.f,0.f,0.f,0.f,0.f,0.f};
    v8f acc1 = (v8f){0.f,0.f,0.f,0.f,0.f,0.f,0.f,0.f};

    #pragma unroll 2
    for (int kt = 0; kt < 128; kt++) {
        int kb = kt*32;
        v8h a0l = *(const v8h*)(rowA0 + kb + ab);
        v8h a0h = *(const v8h*)(rowA0 + kb + ab + 16);
        v8h a1l = *(const v8h*)(rowA1 + kb + ab);
        v8h a1h = *(const v8h*)(rowA1 + kb + ab + 16);
        v8h bl  = *(const v8h*)(rowB  + kb + bb);
        v8h bh  = *(const v8h*)(rowB  + kb + bb + 16);
        v16h a0 = SHUF16(a0l, a0h);
        v16h a1 = SHUF16(a1l, a1h);
        v16h bf = SHUF16(bl,  bh);
        acc0 = __builtin_amdgcn_wmma_f32_16x16x32_f16(false, a0, false, bf, (short)0, acc0, false, false);
        acc1 = __builtin_amdgcn_wmma_f32_16x16x32_f16(false, a1, false, bf, (short)0, acc1, false, false);
    }

    float bb1 = b1[nt*16 + l15];
    #pragma unroll
    for (int r = 0; r < 8; r++) {
        int i0 = mt0*16 + r + (hic ? 8 : 0);
        float x = acc0[r] + bb1;
        h1s[i0][nt*16 + l15] = x > 0.f ? x : 0.01f*x;
        int i1 = mt1*16 + r + (hic ? 8 : 0);
        x = acc1[r] + bb1;
        h1s[i1][nt*16 + l15] = x > 0.f ? x : 0.01f*x;
    }
    __syncthreads();

    for (int idx = tid; idx < 4096; idx += 256) {
        int gl = idx >> 6, j = idx & 63;
        float s = b2[j];
        for (int k = 0; k < 64; k++) s += h1s[gl][k] * W2[k*64 + j];
        h2s[gl][j] = s > 0.f ? s : 0.01f*s;
    }
    __syncthreads();

    if (tid < 64) {
        float s = b3[0];
        for (int k = 0; k < 64; k++) s += h2s[tid][k] * W3[k];
        out[g0 + tid] = s;
    }
}

extern "C" void kernel_launch(void* const* d_in, const int* in_sizes, int n_in,
                              void* d_out, int out_size, void* d_ws, size_t ws_size,
                              hipStream_t stream) {
    (void)in_sizes; (void)n_in; (void)out_size; (void)ws_size;
    const int*   t        = (const int*)  d_in[0];
    const float* c        = (const float*)d_in[1];
    const int*   edges    = (const int*)  d_in[2];
    // d_in[3] batch_idx unused (edges are graph-sorted with fixed EDGES_PER)
    const float* type_emb = (const float*)d_in[4];
    const float* Wp       = (const float*)d_in[5];
    const float* bp       = (const float*)d_in[6];
    const float* Wn       = (const float*)d_in[7];
    const float* bn       = (const float*)d_in[8];
    const float* W1       = (const float*)d_in[9];
    const float* b1       = (const float*)d_in[10];
    const float* W2       = (const float*)d_in[11];
    const float* b2       = (const float*)d_in[12];
    const float* W3       = (const float*)d_in[13];
    const float* b3       = (const float*)d_in[14];
    float* out = (float*)d_out;

    char* ws = (char*)d_ws;
    float*     featF = (float*)ws;                        // 256*4096*4  = 4 MB
    _Float16*  featH = (_Float16*)(ws + (4u << 20));      // 256*4096*2  = 2 MB
    _Float16*  W1T   = (_Float16*)(ws + (6u << 20));      // 64*4096*2   = 512 KB

    k_w1t <<<1024, 256, 0, stream>>>(W1, W1T);
    k_graph<<<NB,   256, 0, stream>>>(t, c, edges, type_emb, Wp, bp, Wn, bn, featF);
    k_cvt <<<4096, 256, 0, stream>>>(featF, featH);
    k_pred<<<4,    256, 0, stream>>>(featH, W1T, b1, W2, b2, W3, b3, out);
}
